// EncoderLayer_2671469658474
// MI455X (gfx1250) — compile-verified, matrix operand order checked by reading, not run on hardware
//
#include <hip/hip_runtime.h>
#include <hip/hip_bf16.h>

#define BB 4
#define SS 2048
#define DDIM 1024
#define HH 16
#define DEPTH 64
#define DFFN 4096
#define NROWS (BB * SS)

typedef __attribute__((ext_vector_type(16))) __bf16 v16bf;
typedef __attribute__((ext_vector_type(8))) float v8f;

// ---------------------------------------------------------------------------
// WMMA fragment helpers (CDNA5 wave32, 16x16x32 bf16)
// A-matrix 16x32 layout (ISA 7.12.2): lane&15 = row (A) / col (B),
// element i holds k = (i>>3)*16 + (lane>>4)*8 + (i&7).
// Source row-major, stride `ld` bf16 elements; k-pairs contiguous -> the
// backend merges the 8 dword loads into global_load_b128s.
// ---------------------------------------------------------------------------
static __device__ __forceinline__ v16bf load_frag_rm(const __bf16* __restrict__ p, int ld) {
  const int lane = threadIdx.x & 31;
  const __bf16* row = p + (size_t)(lane & 15) * ld + ((lane >> 4) << 3);
  v16bf f;
#pragma unroll
  for (int g = 0; g < 2; ++g) {
#pragma unroll
    for (int j = 0; j < 4; ++j) {
      union { uint32_t u; __bf16 h[2]; } c;
      c.u = *reinterpret_cast<const uint32_t*>(row + g * 16 + 2 * j);
      f[g * 8 + 2 * j]     = c.h[0];
      f[g * 8 + 2 * j + 1] = c.h[1];
    }
  }
  return f;
}

static __device__ __forceinline__ v8f wmma_bf16(v16bf a, v16bf b, v8f c) {
  return __builtin_amdgcn_wmma_f32_16x16x32_bf16(false, a, false, b, (short)0, c, false, false);
}

// ---------------------------------------------------------------------------
// Elementwise f32 -> bf16 convert
// ---------------------------------------------------------------------------
__global__ __launch_bounds__(256) void cvt_kernel(const float* __restrict__ x,
                                                  __bf16* __restrict__ y, int n) {
  int i = blockIdx.x * 256 + threadIdx.x;
  if (i < n) y[i] = (__bf16)x[i];
}

// ---------------------------------------------------------------------------
// Weight convert + transpose: W (K x N, f32 row-major) -> Wt (N x K, bf16)
// ---------------------------------------------------------------------------
__global__ __launch_bounds__(256) void wt_kernel(const float* __restrict__ W,
                                                 __bf16* __restrict__ Wt,
                                                 int K, int N) {
  __shared__ float tile[32][33];
  const int bx = blockIdx.x * 32;  // n
  const int by = blockIdx.y * 32;  // k
  const int tx = threadIdx.x & 31;
  const int ty = threadIdx.x >> 5;  // 0..7
#pragma unroll
  for (int i = ty; i < 32; i += 8)
    tile[i][tx] = W[(size_t)(by + i) * N + bx + tx];
  __syncthreads();
#pragma unroll
  for (int i = ty; i < 32; i += 8)
    Wt[(size_t)(bx + i) * K + by + tx] = (__bf16)tile[tx][i];
}

// ---------------------------------------------------------------------------
// bf16 WMMA GEMM: C = A(MxK) * Bt(NxK)^T (+bias)(+resid)(relu)
// Block = 128 threads (4 waves). Block tile 128x64, wave tile 32x64 (2 M-tiles
// x 4 N-tiles = 8 WMMAs per K-step sharing 4 B fragments), K-step 32.
// MODE 0: f32 out (+resid)      MODE 1: bf16 out row-major
// MODE 2: bf16 out (B,H,S,64)   MODE 3: bf16 out (B,H,64,S)
// ---------------------------------------------------------------------------
template <int MODE, bool RELU>
__global__ __launch_bounds__(128) void gemm_kernel(
    const __bf16* __restrict__ A, const __bf16* __restrict__ Bt,
    const float* __restrict__ bias, const float* __restrict__ resid,
    float* __restrict__ outF, __bf16* __restrict__ outB,
    int N, int K) {
  const int wave = threadIdx.x >> 5;
  const int lane = threadIdx.x & 31;
  const int row0 = blockIdx.y * 128 + wave * 32;
  const int col0 = blockIdx.x * 64;

  v8f acc[2][4] = {};
  for (int k0 = 0; k0 < K; k0 += 32) {
    if (k0 + 64 < K) {  // stream-ahead: lowers to global_prefetch_b8
      __builtin_prefetch(A + (size_t)(row0 + lane) * K + k0 + 64, 0, 1);
      __builtin_prefetch(Bt + (size_t)(col0 + lane) * K + k0 + 64, 0, 1);
      __builtin_prefetch(Bt + (size_t)(col0 + 32 + lane) * K + k0 + 64, 0, 1);
    }
    v16bf a0 = load_frag_rm(A + (size_t)row0 * K + k0, K);
    v16bf a1 = load_frag_rm(A + (size_t)(row0 + 16) * K + k0, K);
#pragma unroll
    for (int t = 0; t < 4; ++t) {
      v16bf bf = load_frag_rm(Bt + (size_t)(col0 + 16 * t) * K + k0, K);
      acc[0][t] = wmma_bf16(a0, bf, acc[0][t]);
      acc[1][t] = wmma_bf16(a1, bf, acc[1][t]);
    }
  }

  const int mb = (lane >> 4) << 3;  // row sub-offset (C/D layout)
  const int nl = lane & 15;         // col within 16-tile
#pragma unroll
  for (int mt = 0; mt < 2; ++mt) {
#pragma unroll
    for (int t = 0; t < 4; ++t) {
      const int cg = col0 + 16 * t + nl;
      const float bv = bias[cg];
#pragma unroll
      for (int r = 0; r < 8; ++r) {
        const int rg = row0 + 16 * mt + mb + r;
        float val = acc[mt][t][r] + bv;
        if (RELU) val = fmaxf(val, 0.0f);
        if (MODE == 0) {
          if (resid) val += resid[(size_t)rg * N + cg];
          outF[(size_t)rg * N + cg] = val;
        } else if (MODE == 1) {
          outB[(size_t)rg * N + cg] = (__bf16)val;
        } else {
          const int b = rg / SS, s = rg % SS, h = cg / DEPTH, d = cg % DEPTH;
          if (MODE == 2)
            outB[(((size_t)(b * HH + h) * SS + s) * DEPTH) + d] = (__bf16)val;
          else
            outB[(((size_t)(b * HH + h) * DEPTH + d) * SS) + s] = (__bf16)val;
        }
      }
    }
  }
}

// ---------------------------------------------------------------------------
// Flash attention: one wave = one 32-row query tile (2 WMMA M-tiles),
// streaming 32 keys / iter. K-chunk and V-chunk fragments are reused across
// both query tiles. Row-sum of exp(P) is computed with a WMMA against an
// all-ones B fragment (lands replicated in C/D layout); only the row-max
// needs cross-lane shuffles.
// Q,K in (B,H,S,64) bf16; V in (B,H,64,S) bf16; output attc (B,S,D) bf16.
// ---------------------------------------------------------------------------
__global__ __launch_bounds__(128) void attn_kernel(
    const __bf16* __restrict__ q, const __bf16* __restrict__ kmat,
    const __bf16* __restrict__ vmat, const float* __restrict__ mask,
    __bf16* __restrict__ attc) {
  __shared__ float Pl[4][32 * 32];  // 16 KB
  const int wave = threadIdx.x >> 5;
  const int lane = threadIdx.x & 31;
  const int tid = blockIdx.x * 4 + wave;
  const int tilesPerSeq = SS / 32;
  const int b = tid / (HH * tilesPerSeq);
  const int rem = tid % (HH * tilesPerSeq);
  const int h = rem / tilesPerSeq;
  const int s0 = (rem % tilesPerSeq) * 32;

  const __bf16* Qb = q + ((size_t)(b * HH + h) * SS + s0) * DEPTH;
  const __bf16* Kb = kmat + (size_t)(b * HH + h) * SS * DEPTH;
  const __bf16* Vb = vmat + (size_t)(b * HH + h) * DEPTH * SS;
  const float* mrow = mask + (size_t)b * SS;

  v16bf qf[2][2];
#pragma unroll
  for (int qt = 0; qt < 2; ++qt) {
    qf[qt][0] = load_frag_rm(Qb + (size_t)(16 * qt) * DEPTH, DEPTH);
    qf[qt][1] = load_frag_rm(Qb + (size_t)(16 * qt) * DEPTH + 32, DEPTH);
  }
  v16bf onef;
#pragma unroll
  for (int i = 0; i < 16; ++i) onef[i] = (__bf16)1.0f;

  v8f acc[2][4] = {};
  float rmax[2][8], rsum[2][8];
#pragma unroll
  for (int qt = 0; qt < 2; ++qt)
#pragma unroll
    for (int r = 0; r < 8; ++r) { rmax[qt][r] = -1e30f; rsum[qt][r] = 0.f; }
  const int mb = (lane >> 4) << 3;
  const int nl = lane & 15;
  float* P = Pl[wave];

  for (int kc = 0; kc < SS; kc += 32) {
    if (kc + 32 < SS) {
      __builtin_prefetch(Kb + (size_t)(kc + 32 + lane) * DEPTH, 0, 1);
      __builtin_prefetch(Vb + (size_t)(lane << 1) * SS + kc + 32, 0, 1);
    }
    // shared K-chunk fragments (B-side of QK^T): 32 keys x 64 depth
    v16bf kfa = load_frag_rm(Kb + (size_t)kc * DEPTH, DEPTH);
    v16bf kfb = load_frag_rm(Kb + (size_t)kc * DEPTH + 32, DEPTH);
    v16bf kfc = load_frag_rm(Kb + (size_t)(kc + 16) * DEPTH, DEPTH);
    v16bf kfd = load_frag_rm(Kb + (size_t)(kc + 16) * DEPTH + 32, DEPTH);
    const float m0 = mrow[kc + nl] * -1e9f;
    const float m1 = mrow[kc + 16 + nl] * -1e9f;

    __syncthreads();  // previous iteration's P reads are done
#pragma unroll
    for (int qt = 0; qt < 2; ++qt) {
      v8f sc0 = {}, sc1 = {};
      sc0 = wmma_bf16(qf[qt][0], kfa, sc0);
      sc0 = wmma_bf16(qf[qt][1], kfb, sc0);
      sc1 = wmma_bf16(qf[qt][0], kfc, sc1);
      sc1 = wmma_bf16(qf[qt][1], kfd, sc1);

      float p0[8], p1[8], cm[8];
#pragma unroll
      for (int r = 0; r < 8; ++r) {
        p0[r] = sc0[r] * 0.125f + m0;  // 1/sqrt(64)
        p1[r] = sc1[r] * 0.125f + m1;
        cm[r] = fmaxf(p0[r], p1[r]);
      }
      // row-max over the 16 lanes holding each row (halves independent)
#pragma unroll
      for (int o = 8; o; o >>= 1)
#pragma unroll
        for (int r = 0; r < 8; ++r) cm[r] = fmaxf(cm[r], __shfl_xor(cm[r], o, 32));
#pragma unroll
      for (int r = 0; r < 8; ++r) {
        const float mn = fmaxf(rmax[qt][r], cm[r]);
        const float scl = __expf(rmax[qt][r] - mn);
        rmax[qt][r] = mn;
        rsum[qt][r] *= scl;
        p0[r] = __expf(p0[r] - mn);
        p1[r] = __expf(p1[r] - mn);
#pragma unroll
        for (int t = 0; t < 4; ++t) acc[qt][t][r] *= scl;
        P[(16 * qt + mb + r) * 32 + nl] = p0[r];
        P[(16 * qt + mb + r) * 32 + 16 + nl] = p1[r];
      }
    }
    __syncthreads();

    // rebuild P tiles as A-fragments from LDS
    v16bf pf[2];
#pragma unroll
    for (int qt = 0; qt < 2; ++qt) {
      const float* prow = P + (16 * qt + (lane & 15)) * 32 + ((lane >> 4) << 3);
#pragma unroll
      for (int g = 0; g < 2; ++g)
#pragma unroll
        for (int j = 0; j < 8; ++j) pf[qt][g * 8 + j] = (__bf16)prow[g * 16 + j];
    }
    // row sums of exp(P) via WMMA against ones (replicated across D columns)
    {
      v8f z0 = {}, z1 = {};
      z0 = wmma_bf16(pf[0], onef, z0);
      z1 = wmma_bf16(pf[1], onef, z1);
#pragma unroll
      for (int r = 0; r < 8; ++r) {
        rsum[0][r] += z0[r];
        rsum[1][r] += z1[r];
      }
    }
    // att += P(16x32) * V_chunk(32x64): V fragments shared by both q-tiles
#pragma unroll
    for (int t = 0; t < 4; ++t) {
      v16bf vf = load_frag_rm(Vb + (size_t)(16 * t) * SS + kc, SS);
      acc[0][t] = wmma_bf16(pf[0], vf, acc[0][t]);
      acc[1][t] = wmma_bf16(pf[1], vf, acc[1][t]);
    }
  }

  // normalize and store into (B,S,D) bf16
#pragma unroll
  for (int qt = 0; qt < 2; ++qt) {
#pragma unroll
    for (int t = 0; t < 4; ++t) {
#pragma unroll
      for (int r = 0; r < 8; ++r) {
        const int srow = s0 + 16 * qt + mb + r;
        attc[(size_t)(b * SS + srow) * DDIM + h * DEPTH + 16 * t + nl] =
            (__bf16)(acc[qt][t][r] / rsum[qt][r]);
      }
    }
  }
}

// ---------------------------------------------------------------------------
// LayerNorm: one block per row of D=1024
// ---------------------------------------------------------------------------
__global__ __launch_bounds__(256) void ln_kernel(const float* __restrict__ in,
                                                 const float* __restrict__ g,
                                                 const float* __restrict__ be,
                                                 float* __restrict__ outF,
                                                 __bf16* __restrict__ outB) {
  __shared__ float red[2][8];
  const size_t row = blockIdx.x;
  const float* p = in + row * DDIM;
  float s = 0.f, s2 = 0.f;
  for (int i = threadIdx.x; i < DDIM; i += 256) {
    const float v = p[i];
    s += v;
    s2 += v * v;
  }
#pragma unroll
  for (int o = 16; o; o >>= 1) {
    s += __shfl_xor(s, o, 32);
    s2 += __shfl_xor(s2, o, 32);
  }
  if ((threadIdx.x & 31) == 0) {
    red[0][threadIdx.x >> 5] = s;
    red[1][threadIdx.x >> 5] = s2;
  }
  __syncthreads();
  s = 0.f; s2 = 0.f;
#pragma unroll
  for (int i = 0; i < 8; ++i) { s += red[0][i]; s2 += red[1][i]; }
  const float mu = s * (1.0f / DDIM);
  const float var = s2 * (1.0f / DDIM) - mu * mu;
  const float rs = rsqrtf(var + 1e-6f);
  for (int i = threadIdx.x; i < DDIM; i += 256) {
    const float v = (p[i] - mu) * rs * g[i] + be[i];
    if (outF) outF[row * DDIM + i] = v;
    if (outB) outB[row * DDIM + i] = (__bf16)v;
  }
}

// ---------------------------------------------------------------------------
extern "C" void kernel_launch(void* const* d_in, const int* in_sizes, int n_in,
                              void* d_out, int out_size, void* d_ws, size_t ws_size,
                              hipStream_t stream) {
  (void)in_sizes; (void)n_in; (void)out_size; (void)ws_size;
  const float* x   = (const float*)d_in[0];
  const float* mask= (const float*)d_in[1];
  const float* wq  = (const float*)d_in[2];  const float* bq = (const float*)d_in[3];
  const float* wk  = (const float*)d_in[4];  const float* bk = (const float*)d_in[5];
  const float* wv  = (const float*)d_in[6];  const float* bv = (const float*)d_in[7];
  const float* wo  = (const float*)d_in[8];  const float* bo = (const float*)d_in[9];
  const float* w1  = (const float*)d_in[10]; const float* b1 = (const float*)d_in[11];
  const float* w2  = (const float*)d_in[12]; const float* b2 = (const float*)d_in[13];
  const float* g1  = (const float*)d_in[14]; const float* be1= (const float*)d_in[15];
  const float* g2  = (const float*)d_in[16]; const float* be2= (const float*)d_in[17];
  float* out = (float*)d_out;

  char* ws = (char*)d_ws;
  size_t off = 0;
  auto carve = [&](size_t bytes) -> char* {
    char* p = ws + off;
    off += (bytes + 255) & ~(size_t)255;
    return p;
  };
  __bf16* wqT  = (__bf16*)carve((size_t)DDIM * DDIM * 2);
  __bf16* wkT  = (__bf16*)carve((size_t)DDIM * DDIM * 2);
  __bf16* wvT  = (__bf16*)carve((size_t)DDIM * DDIM * 2);
  __bf16* woT  = (__bf16*)carve((size_t)DDIM * DDIM * 2);
  __bf16* w1T  = (__bf16*)carve((size_t)DDIM * DFFN * 2);
  __bf16* w2T  = (__bf16*)carve((size_t)DFFN * DDIM * 2);
  __bf16* xb   = (__bf16*)carve((size_t)NROWS * DDIM * 2);
  __bf16* qb   = (__bf16*)carve((size_t)NROWS * DDIM * 2);
  __bf16* kb   = (__bf16*)carve((size_t)NROWS * DDIM * 2);
  __bf16* vb   = (__bf16*)carve((size_t)NROWS * DDIM * 2);
  float*  ybuf = (float*) carve((size_t)NROWS * DDIM * 4);
  float*  normf= (float*) carve((size_t)NROWS * DDIM * 4);
  __bf16* hbuf = (__bf16*)carve((size_t)NROWS * DFFN * 2);
  __bf16* attc = xb;    // xb dead after QKV projections
  __bf16* normb= qb;    // qb dead after attention
  float*  zbuf = ybuf;  // ybuf dead after ln1

  // ---- precision conversion ----
  wt_kernel<<<dim3(DDIM / 32, DDIM / 32), 256, 0, stream>>>(wq, wqT, DDIM, DDIM);
  wt_kernel<<<dim3(DDIM / 32, DDIM / 32), 256, 0, stream>>>(wk, wkT, DDIM, DDIM);
  wt_kernel<<<dim3(DDIM / 32, DDIM / 32), 256, 0, stream>>>(wv, wvT, DDIM, DDIM);
  wt_kernel<<<dim3(DDIM / 32, DDIM / 32), 256, 0, stream>>>(wo, woT, DDIM, DDIM);
  wt_kernel<<<dim3(DFFN / 32, DDIM / 32), 256, 0, stream>>>(w1, w1T, DDIM, DFFN);
  wt_kernel<<<dim3(DDIM / 32, DFFN / 32), 256, 0, stream>>>(w2, w2T, DFFN, DDIM);
  cvt_kernel<<<(NROWS * DDIM + 255) / 256, 256, 0, stream>>>(x, xb, NROWS * DDIM);

  const dim3 gD(DDIM / 64, NROWS / 128);
  const dim3 gF(DFFN / 64, NROWS / 128);

  // ---- QKV projections (into head layouts) ----
  gemm_kernel<2, false><<<gD, 128, 0, stream>>>(xb, wqT, bq, nullptr, nullptr, qb, DDIM, DDIM);
  gemm_kernel<2, false><<<gD, 128, 0, stream>>>(xb, wkT, bk, nullptr, nullptr, kb, DDIM, DDIM);
  gemm_kernel<3, false><<<gD, 128, 0, stream>>>(xb, wvT, bv, nullptr, nullptr, vb, DDIM, DDIM);

  // ---- flash attention: B*H*(S/32) query tiles, 4 waves/block ----
  attn_kernel<<<(BB * HH * (SS / 32)) / 4, 128, 0, stream>>>(qb, kb, vb, mask, attc);

  // ---- output projection + residual, layernorm 1 ----
  gemm_kernel<0, false><<<gD, 128, 0, stream>>>(attc, woT, bo, x, ybuf, nullptr, DDIM, DDIM);
  ln_kernel<<<NROWS, 256, 0, stream>>>(ybuf, g1, be1, normf, normb);

  // ---- FFN + residual, layernorm 2 ----
  gemm_kernel<1, true ><<<gF, 128, 0, stream>>>(normb, w1T, b1, nullptr, nullptr, hbuf, DFFN, DDIM);
  gemm_kernel<0, false><<<gD, 128, 0, stream>>>(hbuf, w2T, b2, normf, zbuf, nullptr, DDIM, DFFN);
  ln_kernel<<<NROWS, 256, 0, stream>>>(zbuf, g2, be2, out, nullptr);
}
